// WindowCrossAttn_20942260535578
// MI455X (gfx1250) — matrix operand order repaired and checked by reading, not compiled, not run on hardware
//
#include <hip/hip_runtime.h>
#include <hip/hip_bf16.h>

#define DEV __device__ __forceinline__

typedef __attribute__((ext_vector_type(16))) __bf16 bf16x16;
typedef __attribute__((ext_vector_type(8)))  __bf16 bf16x8;
typedef __attribute__((ext_vector_type(8)))  float  f32x8;

// problem constants
constexpr int DIMM = 180;   // model dim
constexpr int NH   = 6;     // heads
constexpr int HD   = 30;    // head dim
constexpr int HDP  = 32;    // head dim padded
constexpr int NK   = 144;   // keys per window (WIN*WIN)
constexpr int NKP  = 160;   // keys padded to multiple of 32
constexpr int NGSQ = 2304;  // queries per window
constexpr int NB   = 64;    // window batch
constexpr int DP   = 192;   // padded model dim

// workspace layout (bytes)
constexpr size_t OFF_W    = 0;                                   // 4 transposed bf16 weights [192][192]
constexpr size_t SZ_W     = (size_t)4 * DP * DP * 2;             // 294912
constexpr size_t OFF_BIAS = OFF_W + SZ_W;                        // gathered bias [H][NGS][NK] f32
constexpr size_t SZ_BIAS  = (size_t)NH * NGSQ * NK * 4;          // 7962624
constexpr size_t OFF_Q    = OFF_BIAS + SZ_BIAS;                  // q bf16 [B,H,NGS,32]
constexpr size_t SZ_Q     = (size_t)NB * NH * NGSQ * HDP * 2;    // 56623104
constexpr size_t OFF_K    = OFF_Q + SZ_Q;                        // k bf16 [B,H,144,32]
constexpr size_t SZ_K     = (size_t)NB * NH * NK * HDP * 2;      // 3538944
constexpr size_t OFF_V    = OFF_K + SZ_K;                        // v bf16 [B,H,32,160] (hd-major)
constexpr size_t SZ_V     = (size_t)NB * NH * HDP * NKP * 2;     // 3932160
constexpr size_t OFF_X    = OFF_V + SZ_V;                        // attn out bf16 [B*NGS,192]
constexpr size_t SZ_X     = (size_t)NB * NGSQ * DP * 2;          // 56623104
constexpr size_t WS_TOTAL = OFF_X + SZ_X;                        // ~123 MiB

DEV unsigned short f2bf(float f) {
  union { float f; unsigned u; } x; x.f = f;
  unsigned r = x.u + 0x7FFFu + ((x.u >> 16) & 1u);   // round-to-nearest-even
  return (unsigned short)(r >> 16);
}

// ---------------------------------------------------------------------------
// gfx1250 async global->LDS copy (ASYNCcnt path), with synchronous fallback
// ---------------------------------------------------------------------------
#if defined(__has_builtin) && __has_builtin(__builtin_amdgcn_global_load_async_to_lds_b128)
#define HAS_ASYNC_LDS 1
#else
#define HAS_ASYNC_LDS 0
#endif

#if HAS_ASYNC_LDS
typedef int vint4 __attribute__((vector_size(4 * sizeof(int))));
typedef vint4 __attribute__((address_space(1)))* as1_int4p;   // global
typedef vint4 __attribute__((address_space(3)))* as3_int4p;   // LDS
#endif

DEV void async_b128(const void* g, void* l) {   // copies 16 bytes per lane
#if HAS_ASYNC_LDS
  __builtin_amdgcn_global_load_async_to_lds_b128((as1_int4p)g, (as3_int4p)l, 0, 0);
#else
  *(uint4*)l = *(const uint4*)g;
#endif
}

DEV void async_wait0() {
#if HAS_ASYNC_LDS
#if defined(__has_builtin) && __has_builtin(__builtin_amdgcn_s_wait_asynccnt)
  __builtin_amdgcn_s_wait_asynccnt(0);
#else
  asm volatile("s_wait_asynccnt 0x0" ::: "memory");
#endif
#else
  asm volatile("" ::: "memory");
#endif
}

// B-fragment: 16 contiguous bf16 (32 bytes, 32B-aligned)
DEV bf16x16 ld_frag32(const unsigned short* p) {
  return *(const bf16x16*)p;
}
// A-fragment: two contiguous 8-element runs (16B each)
DEV bf16x16 ld_frag_2x16(const unsigned short* lo, const unsigned short* hi) {
  bf16x8 a = *(const bf16x8*)lo;
  bf16x8 b = *(const bf16x8*)hi;
  return __builtin_shufflevector(a, b, 0,1,2,3,4,5,6,7,8,9,10,11,12,13,14,15);
}

DEV f32x8 wmma_bf16(bf16x16 a, bf16x16 b, f32x8 c) {
  return __builtin_amdgcn_wmma_f32_16x16x32_bf16(false, a, false, b,
                                                 (short)0, c, false, false);
}

// ---------------------------------------------------------------------------
// Prep: transpose+pad weights to bf16 (Wq pre-scaled), stored as Wt[n][k]
// ---------------------------------------------------------------------------
__global__ void wca_prep_weights(const float* __restrict__ Wq, const float* __restrict__ Wk,
                                 const float* __restrict__ Wv, const float* __restrict__ Wp,
                                 unsigned short* __restrict__ wt, float scale) {
  int idx = blockIdx.x * blockDim.x + threadIdx.x;
  const int per = DP * DP;
  if (idx >= 4 * per) return;
  int wsel = idx / per, rem = idx % per;
  int n = rem / DP, k = rem % DP;
  const float* W = (wsel == 0) ? Wq : (wsel == 1) ? Wk : (wsel == 2) ? Wv : Wp;
  float v = (n < DIMM && k < DIMM) ? W[k * DIMM + n] : 0.f;
  if (wsel == 0) v *= scale;
  wt[idx] = f2bf(v);
}

// gather bias_table[rel_pos_index] -> biasf[h][q][k]
__global__ void wca_prep_bias(const float* __restrict__ table, const int* __restrict__ rpi,
                              float* __restrict__ biasf) {
  int idx = blockIdx.x * blockDim.x + threadIdx.x;
  if (idx >= NH * NGSQ * NK) return;
  int h = idx / (NGSQ * NK);
  int rem = idx % (NGSQ * NK);           // rem = q*NK + k, matches rpi layout
  biasf[idx] = table[(size_t)rpi[rem] * NH + h];
}

// ---------------------------------------------------------------------------
// Generic 64x192 WMMA GEMM: C = X @ W + bias*bmult, with mode-specific output.
// W streamed in 64-wide K chunks, double-buffered via async global->LDS so the
// next chunk's DMA overlaps the current chunk's 24 WMMAs.
// mode 0: X=gs f32  -> q_ws bf16 [B,H,NGS,32]
// mode 1: X=feat f32-> k_ws bf16 [B,H,144,32]
// mode 2: X=feat f32-> v_ws bf16 [B,H,32,160]   (hd-major)
// mode 3: X=xattn bf16 -> d_out f32 [rows,180]
// ---------------------------------------------------------------------------
__global__ __launch_bounds__(128)
void wca_gemm64x192(const float* __restrict__ Xf, const unsigned short* __restrict__ Xb,
                    const unsigned short* __restrict__ Wt, const float* __restrict__ bias,
                    float bmult, float* __restrict__ outF, unsigned short* __restrict__ outB,
                    int mode) {
  __shared__ alignas(32) unsigned short x_lds[64 * DP];      // 24 KiB, rows bf16
  __shared__ alignas(32) unsigned short w_lds[2][DP * 64];   // 2x24 KiB, Wt chunk [n][64k]

  const int tid = threadIdx.x;
  const int m0 = blockIdx.x * 64;

  // kick off async DMA of W chunk 0 while we stage/convert X
  for (int i = tid; i < DP * 8; i += 128) {
    int n = i >> 3, k4 = i & 7;
    async_b128(Wt + (size_t)n * DP + k4 * 8, &w_lds[0][i * 8]);
  }

  // stage X tile (convert f32->bf16, pad cols 180..191 with zeros)
  if (mode == 3) {
    for (int i = tid; i < 64 * DP / 8; i += 128)
      async_b128(Xb + (size_t)m0 * DP + i * 8, x_lds + i * 8);
  } else {
    for (int i = tid; i < 64 * DP; i += 128) {
      int r = i / DP, c = i % DP;
      float v = (c < DIMM) ? Xf[(size_t)(m0 + r) * DIMM + c] : 0.f;
      x_lds[r * DP + c] = f2bf(v);
    }
  }
  async_wait0();
  __syncthreads();

  const int lane  = tid & 31;
  const int qt    = tid >> 5;               // wave id = 16-row subtile
  const int ln    = lane & 15;
  const int ahalf = (lane & 16) ? 8 : 0;    // C row offset (lanes 16-31 -> +8)
  const int bhalf = (lane & 16) ? 16 : 0;   // B-frag k base (lanes 16-31 -> +16)
  const int mrow  = qt * 16 + ln;

  f32x8 acc[12];
#pragma unroll
  for (int ct = 0; ct < 12; ++ct) acc[ct] = f32x8{0.f,0.f,0.f,0.f,0.f,0.f,0.f,0.f};

  for (int kcc = 0; kcc < 3; ++kcc) {       // K streamed in chunks of 64
    if (kcc < 2) {                          // async-prefetch next chunk into other buffer
      for (int i = tid; i < DP * 8; i += 128) {
        int n = i >> 3, k4 = i & 7;
        async_b128(Wt + (size_t)n * DP + (kcc + 1) * 64 + k4 * 8,
                   &w_lds[(kcc + 1) & 1][i * 8]);
      }
    }
    const unsigned short* wbuf = w_lds[kcc & 1];
#pragma unroll
    for (int kh = 0; kh < 2; ++kh) {
      const unsigned short* ap = x_lds + mrow * DP + kcc * 64 + kh * 32;
      bf16x16 af = ld_frag_2x16(ap + bhalf, ap + bhalf + 8);
#pragma unroll
      for (int ct = 0; ct < 12; ++ct) {
        const unsigned short* bp = wbuf + (ct * 16 + ln) * 64 + kh * 32 + bhalf;
        acc[ct] = wmma_bf16(af, ld_frag32(bp), acc[ct]);
      }
    }
    async_wait0();                          // next chunk's DMA has now landed
    __syncthreads();
  }

  // epilogue
#pragma unroll
  for (int ct = 0; ct < 12; ++ct) {
    int c = ct * 16 + ln;
    if (c >= DIMM) continue;
    float badd = bias[c] * bmult;
#pragma unroll
    for (int r = 0; r < 8; ++r) {
      int row = m0 + qt * 16 + r + ahalf;   // C layout: lanes>=16 -> M+8
      float v = acc[ct][r] + badd;
      if (mode == 0) {
        int b = row / NGSQ, q = row % NGSQ;
        int h = c / HD, j = c % HD;
        outB[((size_t)(b * NH + h) * NGSQ + q) * HDP + j] = f2bf(v);
      } else if (mode == 1) {
        int b = row / NK, kk = row % NK;
        int h = c / HD, j = c % HD;
        outB[((size_t)(b * NH + h) * NK + kk) * HDP + j] = f2bf(v);
      } else if (mode == 2) {
        int b = row / NK, kk = row % NK;
        int h = c / HD, j = c % HD;
        outB[((size_t)(b * NH + h) * HDP + j) * NKP + kk] = f2bf(v);
      } else {
        outF[(size_t)row * DIMM + c] = v;
      }
    }
  }
}

// ---------------------------------------------------------------------------
// Fused attention: one block = 64 q rows x one (b, head). Exact softmax.
// ---------------------------------------------------------------------------
__global__ __launch_bounds__(256)
void wca_attention(const unsigned short* __restrict__ q_ws, const unsigned short* __restrict__ k_ws,
                   const unsigned short* __restrict__ v_ws, const float* __restrict__ biasf,
                   unsigned short* __restrict__ xattn) {
  __shared__ alignas(32) unsigned short q_lds[64 * HDP];   //  4 KiB  [q][hd]
  __shared__ alignas(32) unsigned short k_lds[NK * HDP];   //  9 KiB  [key][hd]
  __shared__ alignas(32) unsigned short v_lds[HDP * NKP];  // 10 KiB  [hd][key]
  __shared__ alignas(32) float          s_lds[64 * NKP];   // 40 KiB  scores; reused as P bf16
  unsigned short* p_lds = (unsigned short*)s_lds;

  const int tid = threadIdx.x;
  int bid = blockIdx.x;
  const int qtile = bid % (NGSQ / 64); bid /= (NGSQ / 64);
  const int h = bid % NH;
  const int b = bid / NH;
  const int q0 = qtile * 64;

  const unsigned short* qg = q_ws + ((size_t)(b * NH + h) * NGSQ + q0) * HDP;
  const unsigned short* kg = k_ws + (size_t)(b * NH + h) * NK * HDP;
  const unsigned short* vg = v_ws + (size_t)(b * NH + h) * HDP * NKP;
  const float* bg = biasf + ((size_t)h * NGSQ + q0) * NK;

  // async DMA of Q/K/V tiles into LDS (16B per lane per issue)
  for (int i = tid; i < 64 * HDP / 8; i += 256) async_b128(qg + i * 8, q_lds + i * 8);
  for (int i = tid; i < NK * HDP / 8; i += 256) async_b128(kg + i * 8, k_lds + i * 8);
  for (int i = tid; i < HDP * NKP / 8; i += 256) async_b128(vg + i * 8, v_lds + i * 8);
  __builtin_prefetch(bg + (size_t)tid * 36, 0, 0);   // warm bias rows for softmax
  async_wait0();
  __syncthreads();

  const int lane  = tid & 31;
  const int wid   = tid >> 5;
  const int ln    = lane & 15;
  const int ahalf = (lane & 16) ? 8 : 0;
  const int bhalf = (lane & 16) ? 16 : 0;

  // S = Q @ K^T : 4 q-subtiles x 9 key-tiles = 36 WMMA tiles over 8 waves
  for (int t = wid; t < 36; t += 8) {
    int sqt = t / 9, kt = t % 9;
    const unsigned short* ap = q_lds + (sqt * 16 + ln) * HDP;
    bf16x16 af = ld_frag_2x16(ap + bhalf, ap + bhalf + 8);
    const unsigned short* bp = k_lds + (kt * 16 + ln) * HDP + bhalf;
    f32x8 acc = {0.f,0.f,0.f,0.f,0.f,0.f,0.f,0.f};
    acc = wmma_bf16(af, ld_frag32(bp), acc);
    int scol = kt * 16 + ln;
#pragma unroll
    for (int r = 0; r < 8; ++r)
      s_lds[(sqt * 16 + r + ahalf) * NKP + scol] = acc[r];
  }
  __syncthreads();

  // softmax over 144 keys: 4 lanes per row (64 rows * 4 = 256 threads)
  {
    const int r = tid >> 2;
    const int pp = tid & 3;
    float a[36];
    const float* brow = bg + (size_t)r * NK + pp * 36;
    float mx = -3.0e38f;
#pragma unroll
    for (int i = 0; i < 36; ++i) {
      a[i] = s_lds[r * NKP + pp * 36 + i] + brow[i];
      mx = fmaxf(mx, a[i]);
    }
    mx = fmaxf(mx, __shfl_xor(mx, 1, 32));
    mx = fmaxf(mx, __shfl_xor(mx, 2, 32));
    float sum = 0.f;
#pragma unroll
    for (int i = 0; i < 36; ++i) { a[i] = __expf(a[i] - mx); sum += a[i]; }
    sum += __shfl_xor(sum, 1, 32);
    sum += __shfl_xor(sum, 2, 32);
    float inv = 1.0f / sum;
    __syncthreads();                       // all S reads complete before P overwrite
#pragma unroll
    for (int i = 0; i < 36; ++i)
      p_lds[r * NKP + pp * 36 + i] = f2bf(a[i] * inv);
    if (pp == 3) {                         // zero key-padding 144..159
#pragma unroll
      for (int i = 0; i < 16; ++i) p_lds[r * NKP + NK + i] = 0;
    }
  }
  __syncthreads();

  // O = P @ V : 4 q-subtiles x 2 hd-tiles, K-dim = 160 keys in 5 chunks
  {
    const int sqt = wid >> 1, ht = wid & 1;
    f32x8 acc = {0.f,0.f,0.f,0.f,0.f,0.f,0.f,0.f};
#pragma unroll
    for (int kc = 0; kc < 5; ++kc) {
      const unsigned short* ap = p_lds + (sqt * 16 + ln) * NKP + kc * 32;
      bf16x16 af = ld_frag_2x16(ap + bhalf, ap + bhalf + 8);
      const unsigned short* bp = v_lds + (ht * 16 + ln) * NKP + kc * 32 + bhalf;
      acc = wmma_bf16(af, ld_frag32(bp), acc);
    }
    int j = ht * 16 + ln;
    if (j < HD) {
#pragma unroll
      for (int r = 0; r < 8; ++r) {
        int row = q0 + sqt * 16 + r + ahalf;
        xattn[((size_t)b * NGSQ + row) * DP + h * HD + j] = f2bf(acc[r]);
      }
    }
  }
}

// ---------------------------------------------------------------------------
extern "C" void kernel_launch(void* const* d_in, const int* in_sizes, int n_in,
                              void* d_out, int out_size, void* d_ws, size_t ws_size,
                              hipStream_t stream) {
  (void)in_sizes; (void)n_in; (void)out_size;
  const float* gs    = (const float*)d_in[0];
  const float* feat  = (const float*)d_in[1];
  const float* Wq    = (const float*)d_in[2];
  const float* bq    = (const float*)d_in[3];
  const float* Wk    = (const float*)d_in[4];
  const float* bk    = (const float*)d_in[5];
  const float* Wv    = (const float*)d_in[6];
  const float* bv    = (const float*)d_in[7];
  const float* Wp    = (const float*)d_in[8];
  const float* bp    = (const float*)d_in[9];
  const float* table = (const float*)d_in[10];
  const int*   rpi   = (const int*)d_in[11];

  char* ws = (char*)d_ws;
  unsigned short* wt    = (unsigned short*)(ws + OFF_W);
  float*          biasf = (float*)(ws + OFF_BIAS);
  unsigned short* q_ws  = (unsigned short*)(ws + OFF_Q);
  unsigned short* k_ws  = (unsigned short*)(ws + OFF_K);
  unsigned short* v_ws  = (unsigned short*)(ws + OFF_V);
  unsigned short* xatt  = (unsigned short*)(ws + OFF_X);

  const float scale = 0.18257418583505536f;  // 30^-0.5

  size_t clr = WS_TOTAL < ws_size ? WS_TOTAL : ws_size;
  (void)hipMemsetAsync(d_ws, 0, clr, stream);  // establishes all zero-padding

  wca_prep_weights<<<(4 * DP * DP + 255) / 256, 256, 0, stream>>>(Wq, Wk, Wv, Wp, wt, scale);
  wca_prep_bias<<<(NH * NGSQ * NK + 255) / 256, 256, 0, stream>>>(table, rpi, biasf);

  // Q/K/V projections (bias of Q also scaled via bmult)
  wca_gemm64x192<<<NB * NGSQ / 64, 128, 0, stream>>>(gs, nullptr, wt + 0 * DP * DP, bq, scale,
                                                     nullptr, q_ws, 0);
  wca_gemm64x192<<<NB * NK / 64, 128, 0, stream>>>(feat, nullptr, wt + 1 * DP * DP, bk, 1.f,
                                                   nullptr, k_ws, 1);
  wca_gemm64x192<<<NB * NK / 64, 128, 0, stream>>>(feat, nullptr, wt + 2 * DP * DP, bv, 1.f,
                                                   nullptr, v_ws, 2);

  wca_attention<<<NB * NH * (NGSQ / 64), 256, 0, stream>>>(q_ws, k_ws, v_ws, biasf, xatt);

  // output projection -> fp32 d_out
  wca_gemm64x192<<<NB * NGSQ / 64, 128, 0, stream>>>(nullptr, xatt, wt + 3 * DP * DP, bp, 1.f,
                                                     (float*)d_out, nullptr, 3);
}
